// DFGI_54382875902481
// MI455X (gfx1250) — matrix operand order repaired and checked by reading, not compiled, not run on hardware
//
#include <hip/hip_runtime.h>
#include <hip/hip_bf16.h>
#include <cstdint>
#include <cstddef>

// ---------------------------------------------------------------------------
// CDNA5 (gfx1250, wave32) WMMA helpers
// Fragment layouts used here (A/B: this file's contiguous k order, same for both; C/D per §7.12.2):
//  A (16x32 f16): lane m = lane&15 (both halves), element e -> k =
//                 (lane>>4)*16 + e
//  B (32x16 f16): lane n = lane&15, element e -> k = (lane>>4)*16 + e
//  C/D (16x16 f32): row m = (lane>>4)*8 + v, col n = lane&15
// ---------------------------------------------------------------------------
typedef __attribute__((ext_vector_type(16))) _Float16 v16h;
typedef __attribute__((ext_vector_type(8)))  float    v8f;

__device__ __forceinline__ v8f wmma16x16x32(v16h a, v16h b, v8f c) {
  return __builtin_amdgcn_wmma_f32_16x16x32_f16(
      /*neg_a=*/false, a, /*neg_b=*/false, b,
      /*c_mod=*/(short)0, c, /*reuse_a=*/false, /*reuse_b=*/false);
}

// ---------------------------------------------------------------------------
// Pack conv weights [OC][256][kh][kw] f32 -> [KK][OC][256] f16
// ---------------------------------------------------------------------------
__global__ void pack_w_kernel(const float* __restrict__ w, _Float16* __restrict__ wp,
                              int OC, int KK) {
  int idx = blockIdx.x * blockDim.x + threadIdx.x;
  int total = OC * 256 * KK;
  if (idx >= total) return;
  int kk = idx % KK;
  int c  = (idx / KK) % 256;
  int oc = idx / (KK * 256);
  wp[((size_t)kk * OC + oc) * 256 + c] = (_Float16)w[idx];
}

// ---------------------------------------------------------------------------
// Bilinear (align_corners) resize [B][256][H][W] f32 -> padded CHANNEL-LAST
// f16 [B][34][34][256] (zero ring for SAME 3x3 conv; inner 32x32 = resized).
// Channel-last makes the conv's im2col B-fragment a single contiguous 32-byte
// run per lane (global_load_b128 instead of 16 channel-strided d16 gathers).
// ---------------------------------------------------------------------------
__global__ void resize_pad_f16(const float* __restrict__ in, _Float16* __restrict__ outp,
                               int B, int H, int W) {
  size_t idx = (size_t)blockIdx.x * blockDim.x + threadIdx.x;
  size_t total = (size_t)B * 256 * 34 * 34;
  if (idx >= total) return;
  // outp index = ((b*34 + y)*34 + x)*256 + c  (== idx)
  int c = (int)(idx % 256);
  size_t t = idx / 256;
  int x = (int)(t % 34); t /= 34;
  int y = (int)(t % 34);
  int b = (int)(t / 34);
  float val = 0.0f;
  if (x >= 1 && x <= 32 && y >= 1 && y <= 32) {
    int oy = y - 1, ox = x - 1;
    float fy = oy * ((float)(H - 1) / 31.0f);
    float fx = ox * ((float)(W - 1) / 31.0f);
    int y0 = (int)floorf(fy), x0 = (int)floorf(fx);
    int y1 = min(y0 + 1, H - 1), x1 = min(x0 + 1, W - 1);
    float wy = fy - (float)y0, wx = fx - (float)x0;
    const float* p = in + ((size_t)(b * 256 + c) * H) * W;
    float v00 = p[y0 * W + x0], v01 = p[y0 * W + x1];
    float v10 = p[y1 * W + x0], v11 = p[y1 * W + x1];
    val = (v00 * (1.f - wx) + v01 * wx) * (1.f - wy) +
          (v10 * (1.f - wx) + v11 * wx) * wy;
  }
  outp[idx] = (_Float16)val;
}

// ---------------------------------------------------------------------------
// Implicit-GEMM conv via WMMA. One wave per 16(oc) x 16(pos) tile.
//   frp : [NIMG][34][34][256] f16 (padded, channel-last)
//   Wp  : [KK][OC][256] f16
//   modes: 0 -> f16 out [img][OC][1024]
//          1 -> f16 out [img][1024][OC]   (row-per-position, A/B-frag friendly)
//          2 -> f32 out [img][OC][1024]
// ---------------------------------------------------------------------------
__global__ void conv_gemm_wmma(const _Float16* __restrict__ frp,
                               const _Float16* __restrict__ Wp,
                               const float* __restrict__ bias,
                               void* __restrict__ out,
                               int OC, int KK, int mode) {
  const int lane = threadIdx.x;
  const int col  = lane & 15;
  const int half = lane >> 4;
  const int ntile  = blockIdx.x;   // 64 tiles of 16 positions
  const int octile = blockIdx.y;   // OC/16
  const int img    = blockIdx.z;

  const int pos = ntile * 16 + col;
  const int py  = pos >> 5;
  const int px  = pos & 31;

  v8f acc = {};
  for (int kk = 0; kk < KK; ++kk) {
    const int ky = (KK == 9) ? (kk / 3) : 1;
    const int kx = (KK == 9) ? (kk % 3) : 1;
    // channel-last: all 256 channels of pixel (py+ky, px+kx) are contiguous
    const _Float16* fbase =
        frp + (((size_t)img * 34 + (py + ky)) * 34 + (px + kx)) * 256;
    const _Float16* wbase = Wp + ((size_t)kk * OC + octile * 16 + col) * 256;
    for (int c0 = 0; c0 < 256; c0 += 32) {
      v16h a, b;
#pragma unroll
      for (int e = 0; e < 16; ++e)
        a[e] = wbase[c0 + half * 16 + e];
#pragma unroll
      for (int e = 0; e < 16; ++e)
        b[e] = fbase[c0 + half * 16 + e];          // one contiguous 32B run
      acc = wmma16x16x32(a, b, acc);
    }
  }

  if (mode == 0) {
    _Float16* o = (_Float16*)out;
#pragma unroll
    for (int v = 0; v < 8; ++v) {
      int oc = octile * 16 + half * 8 + v;
      o[((size_t)img * OC + oc) * 1024 + pos] = (_Float16)(acc[v] + bias[oc]);
    }
  } else if (mode == 1) {
    _Float16* o = (_Float16*)out;
#pragma unroll
    for (int v = 0; v < 8; ++v) {
      int oc = octile * 16 + half * 8 + v;
      o[((size_t)img * 1024 + pos) * OC + oc] = (_Float16)(acc[v] + bias[oc]);
    }
  } else {
    float* o = (float*)out;
#pragma unroll
    for (int v = 0; v < 8; ++v) {
      int oc = octile * 16 + half * 8 + v;
      o[((size_t)img * OC + oc) * 1024 + pos] = acc[v] + bias[oc];
    }
  }
}

// ---------------------------------------------------------------------------
// gap_key[n][32] = conv1x1(mean_hw(supFeatures[n]), W) + b
// ---------------------------------------------------------------------------
__global__ void gap_key_kernel(const float* __restrict__ sup,
                               const float* __restrict__ w,
                               const float* __restrict__ b,
                               float* __restrict__ gk) {
  __shared__ float mean_s[256];
  int n = blockIdx.x;
  int t = threadIdx.x;
  const float* p = sup + ((size_t)n * 256 + t) * 1024;
  float s = 0.f;
  for (int i = 0; i < 1024; ++i) s += p[i];
  mean_s[t] = s * (1.0f / 1024.0f);
  __syncthreads();
  if (t < 32) {
    float acc = b[t];
    const float* wr = w + t * 256;
    for (int c = 0; c < 256; ++c) acc += wr[c] * mean_s[c];
    gk[n * 32 + t] = acc;
  }
}

// ---------------------------------------------------------------------------
// atten[b][q] = sum_n sigmoid( kq[b][q][:] . gap_key[n][:] )
// ---------------------------------------------------------------------------
__global__ void gating_kernel(const _Float16* __restrict__ kqh,
                              const float* __restrict__ gk,
                              float* __restrict__ atten) {
  int idx = blockIdx.x * blockDim.x + threadIdx.x;
  if (idx >= 4096) return;
  const _Float16* row = kqh + (size_t)idx * 32;
  float kr[32];
#pragma unroll
  for (int c = 0; c < 32; ++c) kr[c] = (float)row[c];
  float s = 0.f;
  for (int n = 0; n < 5; ++n) {
    float d = 0.f;
#pragma unroll
    for (int c = 0; c < 32; ++c) d += kr[c] * gk[n * 32 + c];
    s += 1.0f / (1.0f + __expf(-d));
  }
  atten[idx] = s;
}

// ---------------------------------------------------------------------------
// Flash-style cross attention, one wave per (s-tile16, batch, class).
//   logits[s,q] = supk[n][s][:] . kq[b][q][:]   (K=32 -> one WMMA / 16x16 tile)
//   softmax over q (online, 32-wide chunks), O[s,c128] = P.V
//   val[b][c][s] += O / l   (atomic sum over classes n)
// supk: [5][1024][32] f16, supv: [5][128][1024] f16, kqh: [4][1024][32] f16
// ---------------------------------------------------------------------------
__global__ void attention_wmma(const _Float16* __restrict__ supk,
                               const _Float16* __restrict__ supv,
                               const _Float16* __restrict__ kqh,
                               float* __restrict__ val) {
  __shared__ float    Ls[16 * 32];   // logits [s][q]
  __shared__ _Float16 Ps[16 * 32];   // probs  [s][q]
  __shared__ float    Sc[16];        // per-row rescale
  __shared__ float    Il[16];        // per-row 1/l

  const int lane = threadIdx.x;
  const int col  = lane & 15;
  const int half = lane >> 4;
  const int s0 = blockIdx.x * 16;
  const int b  = blockIdx.y;
  const int n  = blockIdx.z;

  // A fragment: sup_key rows s0..s0+15 (fixed for this block)
  v16h aK;
  const _Float16* ar = supk + ((size_t)n * 1024 + s0 + col) * 32;
#pragma unroll
  for (int e = 0; e < 16; ++e)
    aK[e] = ar[half * 16 + e];

  float o[8][8];
#pragma unroll
  for (int t = 0; t < 8; ++t)
#pragma unroll
    for (int v = 0; v < 8; ++v) o[t][v] = 0.f;
  float m = -1e30f, l = 0.f;   // running stats (valid on lanes 0..15, row=lane)

  for (int q0 = 0; q0 < 1024; q0 += 32) {
    // --- logits tile [16 s][32 q]: two WMMAs ---
#pragma unroll
    for (int st = 0; st < 2; ++st) {
      v16h bQ;
      const _Float16* br = kqh + ((size_t)b * 1024 + q0 + st * 16 + col) * 32;
      __builtin_prefetch(br + 32 * 32, 0, 0);  // global_prefetch_b8 next chunk
#pragma unroll
      for (int e = 0; e < 16; ++e) bQ[e] = br[half * 16 + e];
      v8f c = {};
      c = wmma16x16x32(aK, bQ, c);
#pragma unroll
      for (int v = 0; v < 8; ++v)
        Ls[(half * 8 + v) * 32 + st * 16 + col] = c[v];
    }
    __syncthreads();

    // --- online softmax rows (lanes 0..15, one s-row each) ---
    if (lane < 16) {
      const float* lr = Ls + lane * 32;
      float mx = m;
      for (int q = 0; q < 32; ++q) mx = fmaxf(mx, lr[q]);
      float corr = __expf(m - mx);
      float s = l * corr;
      _Float16* pr = Ps + lane * 32;
      for (int q = 0; q < 32; ++q) {
        float p = __expf(lr[q] - mx);
        s += p;
        pr[q] = (_Float16)p;
      }
      m = mx; l = s; Sc[lane] = corr;
    }
    __syncthreads();

    // --- rescale accumulator + O += P.V (8 WMMAs) ---
    float sc[8];
#pragma unroll
    for (int v = 0; v < 8; ++v) sc[v] = Sc[half * 8 + v];
    v16h aP;
    const _Float16* prow = Ps + col * 32;
#pragma unroll
    for (int e = 0; e < 16; ++e)
      aP[e] = prow[half * 16 + e];
#pragma unroll
    for (int ct = 0; ct < 8; ++ct) {
      v16h bV;
      const _Float16* vr =
          supv + ((size_t)n * 128 + ct * 16 + col) * 1024 + q0 + half * 16;
#pragma unroll
      for (int e = 0; e < 16; ++e) bV[e] = vr[e];
      v8f c;
#pragma unroll
      for (int v = 0; v < 8; ++v) c[v] = o[ct][v] * sc[v];
      c = wmma16x16x32(aP, bV, c);
#pragma unroll
      for (int v = 0; v < 8; ++v) o[ct][v] = c[v];
    }
    __syncthreads();
  }

  if (lane < 16) Il[lane] = 1.0f / l;
  __syncthreads();
  float il[8];
#pragma unroll
  for (int v = 0; v < 8; ++v) il[v] = Il[half * 8 + v];
#pragma unroll
  for (int ct = 0; ct < 8; ++ct)
#pragma unroll
    for (int v = 0; v < 8; ++v) {
      int c = ct * 16 + col;          // C-frag col = value channel
      int s = s0 + half * 8 + v;      // C-frag row = s position
      atomicAdd(&val[((size_t)b * 128 + c) * 1024 + s], o[ct][v] * il[v]);
    }
}

// ---------------------------------------------------------------------------
// out[b][ch][H][W] = BN( resize_ac_{32->H,W}( ch<128 ? vq*atten : val ) )
// ---------------------------------------------------------------------------
__global__ void combine_resize_bn(const float* __restrict__ vq,
                                  const float* __restrict__ val,
                                  const float* __restrict__ atten,
                                  const float* __restrict__ g,
                                  const float* __restrict__ bb,
                                  const float* __restrict__ mm,
                                  const float* __restrict__ vv,
                                  float* __restrict__ out, int H, int W) {
  size_t idx = (size_t)blockIdx.x * blockDim.x + threadIdx.x;
  size_t total = (size_t)4 * 256 * H * W;
  if (idx >= total) return;
  int x = (int)(idx % W);
  size_t t = idx / W;
  int y = (int)(t % H); t /= H;
  int ch = (int)(t % 256);
  int b  = (int)(t / 256);

  float fy = y * (31.0f / (float)(H - 1));
  float fx = x * (31.0f / (float)(W - 1));
  int y0 = (int)floorf(fy), x0 = (int)floorf(fx);
  int y1 = min(y0 + 1, 31), x1 = min(x0 + 1, 31);
  float wy = fy - (float)y0, wx = fx - (float)x0;

  float s00, s01, s10, s11;
  if (ch < 128) {
    const float* src = vq + ((size_t)b * 128 + ch) * 1024;
    const float* at  = atten + (size_t)b * 1024;
    s00 = src[y0 * 32 + x0] * at[y0 * 32 + x0];
    s01 = src[y0 * 32 + x1] * at[y0 * 32 + x1];
    s10 = src[y1 * 32 + x0] * at[y1 * 32 + x0];
    s11 = src[y1 * 32 + x1] * at[y1 * 32 + x1];
  } else {
    const float* src = val + ((size_t)b * 128 + (ch - 128)) * 1024;
    s00 = src[y0 * 32 + x0]; s01 = src[y0 * 32 + x1];
    s10 = src[y1 * 32 + x0]; s11 = src[y1 * 32 + x1];
  }
  float r = (s00 * (1.f - wx) + s01 * wx) * (1.f - wy) +
            (s10 * (1.f - wx) + s11 * wx) * wy;
  float scale = g[ch] * rsqrtf(vv[ch] + 1e-5f);
  out[idx] = (r - mm[ch]) * scale + bb[ch];
}

// ---------------------------------------------------------------------------
extern "C" void kernel_launch(void* const* d_in, const int* in_sizes, int n_in,
                              void* d_out, int out_size, void* d_ws, size_t ws_size,
                              hipStream_t stream) {
  (void)in_sizes; (void)n_in; (void)out_size; (void)ws_size;

  const float* feats[5] = {(const float*)d_in[0], (const float*)d_in[1],
                           (const float*)d_in[2], (const float*)d_in[3],
                           (const float*)d_in[4]};
  const int HWs[5][2] = {{64, 64}, {32, 32}, {16, 16}, {8, 8}, {8, 8}};
  const float* supF = (const float*)d_in[5];
  const float* gapW = (const float*)d_in[6];
  const float* gapB = (const float*)d_in[7];
  const float* svW  = (const float*)d_in[8];
  const float* svB  = (const float*)d_in[9];
  const float* skW  = (const float*)d_in[10];
  const float* skB  = (const float*)d_in[11];

  char* ws = (char*)d_ws;
  size_t off = 0;
  auto wsAlloc = [&](size_t bytes) -> void* {
    void* p = ws + off;
    off = (off + bytes + 255) & ~(size_t)255;
    return p;
  };
  _Float16* supPad = (_Float16*)wsAlloc((size_t)5 * 256 * 34 * 34 * 2);
  _Float16* frPad  = (_Float16*)wsAlloc((size_t)4 * 256 * 34 * 34 * 2);
  _Float16* WpSV   = (_Float16*)wsAlloc((size_t)9 * 128 * 256 * 2);
  _Float16* WpSK   = (_Float16*)wsAlloc((size_t)9 * 32 * 256 * 2);
  _Float16* WpV    = (_Float16*)wsAlloc((size_t)9 * 128 * 256 * 2);
  _Float16* WpK    = (_Float16*)wsAlloc((size_t)9 * 32 * 256 * 2);
  _Float16* supvH  = (_Float16*)wsAlloc((size_t)5 * 128 * 1024 * 2);
  _Float16* supkH  = (_Float16*)wsAlloc((size_t)5 * 1024 * 32 * 2);
  float*    gapKey = (float*)wsAlloc((size_t)5 * 32 * 4);
  _Float16* kqH    = (_Float16*)wsAlloc((size_t)4 * 1024 * 32 * 2);
  float*    vqF    = (float*)wsAlloc((size_t)4 * 128 * 1024 * 4);
  float*    valF   = (float*)wsAlloc((size_t)4 * 128 * 1024 * 4);
  float*    atten  = (float*)wsAlloc((size_t)4 * 1024 * 4);

  // ---- support side (once per launch) ----
  {
    int tot = 5 * 256 * 34 * 34;
    resize_pad_f16<<<(tot + 255) / 256, 256, 0, stream>>>(supF, supPad, 5, 32, 32);
  }
  {
    int tot = 128 * 256 * 9;
    pack_w_kernel<<<(tot + 255) / 256, 256, 0, stream>>>(svW, WpSV, 128, 9);
  }
  {
    int tot = 32 * 256 * 9;
    pack_w_kernel<<<(tot + 255) / 256, 256, 0, stream>>>(skW, WpSK, 32, 9);
  }
  conv_gemm_wmma<<<dim3(64, 8, 5), 32, 0, stream>>>(supPad, WpSV, svB, supvH, 128, 9, 0);
  conv_gemm_wmma<<<dim3(64, 2, 5), 32, 0, stream>>>(supPad, WpSK, skB, supkH, 32, 9, 1);
  gap_key_kernel<<<5, 256, 0, stream>>>(supF, gapW, gapB, gapKey);

  // ---- per-scale pipeline ----
  size_t outOff = 0;
  for (int i = 0; i < 5; ++i) {
    const int H = HWs[i][0], W = HWs[i][1];
    const int KK = (i < 4) ? 9 : 1;
    const float* qkW = (const float*)d_in[12 + 8 * i];
    const float* qkB = (const float*)d_in[13 + 8 * i];
    const float* qvW = (const float*)d_in[14 + 8 * i];
    const float* qvB = (const float*)d_in[15 + 8 * i];
    const float* bnG = (const float*)d_in[16 + 8 * i];
    const float* bnB = (const float*)d_in[17 + 8 * i];
    const float* bnM = (const float*)d_in[18 + 8 * i];
    const float* bnV = (const float*)d_in[19 + 8 * i];

    {
      int tot = 4 * 256 * 34 * 34;
      resize_pad_f16<<<(tot + 255) / 256, 256, 0, stream>>>(feats[i], frPad, 4, H, W);
    }
    {
      int tot = 32 * 256 * KK;
      pack_w_kernel<<<(tot + 255) / 256, 256, 0, stream>>>(qkW, WpK, 32, KK);
    }
    {
      int tot = 128 * 256 * KK;
      pack_w_kernel<<<(tot + 255) / 256, 256, 0, stream>>>(qvW, WpV, 128, KK);
    }
    conv_gemm_wmma<<<dim3(64, 2, 4), 32, 0, stream>>>(frPad, WpK, qkB, kqH, 32, KK, 1);
    conv_gemm_wmma<<<dim3(64, 8, 4), 32, 0, stream>>>(frPad, WpV, qvB, vqF, 128, KK, 2);
    gating_kernel<<<16, 256, 0, stream>>>(kqH, gapKey, atten);
    hipMemsetAsync(valF, 0, (size_t)4 * 128 * 1024 * 4, stream);
    attention_wmma<<<dim3(64, 4, 5), 32, 0, stream>>>(supkH, supvH, kqH, valF);
    {
      size_t tot = (size_t)4 * 256 * H * W;
      combine_resize_bn<<<(unsigned)((tot + 255) / 256), 256, 0, stream>>>(
          vqF, valF, atten, bnG, bnB, bnM, bnV, (float*)d_out + outOff, H, W);
      outOff += tot;
    }
  }
}
